// Controller_37924561223944
// MI455X (gfx1250) — matrix operand order repaired and checked by reading, not compiled, not run on hardware
//
#include <hip/hip_runtime.h>

typedef __attribute__((ext_vector_type(16))) _Float16 v16h;
typedef __attribute__((ext_vector_type(8)))  float    v8f;
typedef __attribute__((ext_vector_type(2)))  _Float16 h2;

union Frag16 { v16h v; h2 p[8]; };

__device__ __forceinline__ float relu32(float x) {
  return __builtin_amdgcn_fmed3f(x, 0.0f, __builtin_inff());
}

// 8 tiny MLPs (H=64) over B samples, fused, TRANSPOSED layer-2 GEMM:
//   H2^T[64 x 16] = W2^T @ H1^T  per 16-sample chunk  (v_wmma_f32_16x16x32_f16)
//   -> D holds channels along M (lanes own rows), samples along N (lanes)
//   -> layer-3 dot w3 is per-lane accumulation + ONE shfl_xor(16) merge
// Block = 256 threads = 8 wave32; wave w owns net w.
// GRID-STRIDE over 256-sample tiles: weights stay in registers across tiles,
// amortizing the ~22.5KB/wave fragment preload over ~1000 samples instead of 256.
__global__ __launch_bounds__(256, 1) void mlp8_wmma_kernel(
    const float* __restrict__ s,  const float* __restrict__ w1,
    const float* __restrict__ b1, const float* __restrict__ w2,
    const float* __restrict__ b2, const float* __restrict__ w3,
    const float* __restrict__ b3, float* __restrict__ out, int B)
{
  __shared__ float mbuf[8][256];

  const int tid  = threadIdx.x;
  const int lane = tid & 31;
  const int net  = tid >> 5;        // one net per wave
  const int col  = lane & 15;
  const int half = lane >> 4;
  const int sidx = net & 3;         // STATE_IDX = [0,1,2,3,0,1,2,3]

  const h2 hzero = {(_Float16)0.0f, (_Float16)0.0f};

  // ---- layer-1 weights in B-fragment order (H1^T is the B matrix) ----
  // B 32x16 f16 layout: lane col = sample, element e -> k = 32*kt + 8*half + e (+8 if e>=8)
  h2 w1p[16], b1p[16];
#pragma unroll
  for (int kt = 0; kt < 2; ++kt)
#pragma unroll
    for (int i = 0; i < 8; ++i) {
      int kb = 32*kt + 8*half + 2*i + (i >= 4 ? 8 : 0);
      const float* wp = w1 + net*64 + kb;
      const float* bp = b1 + net*64 + kb;
      w1p[kt*8+i] = h2{(_Float16)wp[0], (_Float16)wp[1]};
      b1p[kt*8+i] = h2{(_Float16)bp[0], (_Float16)bp[1]};
    }

  // ---- W2^T as A fragments: A[c][h] = w2[net][h][c] ----
  // A 16x32 f16 layout: row M = col; element e -> h = 32*kt + 8*half + e (+8 if e>=8)
  Frag16 Af[2][4];
#pragma unroll
  for (int kt = 0; kt < 2; ++kt)
#pragma unroll
    for (int mt = 0; mt < 4; ++mt)
#pragma unroll
      for (int i = 0; i < 8; ++i) {
        int h0 = 32*kt + 8*half + 2*i + (i >= 4 ? 8 : 0);
        int cch = 16*mt + col;
        const float* p = w2 + net*4096 + h0*64 + cch;
        Af[kt][mt].p[i] = h2{(_Float16)p[0], (_Float16)p[64]};
      }

  // ---- per-lane b2 / w3 tables: D element (mt, r) is channel 16*mt + 8*half + r ----
  float b2v[4][8], w3v[4][8];
#pragma unroll
  for (int mt = 0; mt < 4; ++mt)
#pragma unroll
    for (int r = 0; r < 8; ++r) {
      int idx = net*64 + 16*mt + 8*half + r;
      b2v[mt][r] = b2[idx];
      w3v[mt][r] = w3[idx];
    }
  const float b3v = b3[net];

  // ---- grid-stride over 256-sample tiles (uniform across block: barriers safe) ----
  for (int base = blockIdx.x * 256; base < B; base += gridDim.x * 256) {

    // 16 chunks of 16 samples
    for (int c = 0; c < 16; ++c) {
      int row = base + c*16 + col;
      if (row > B - 1) row = B - 1;          // clamp tail loads
      float xs = s[row*4 + sidx];
      if (c < 15)                             // gfx1250 global_prefetch_b8
        __builtin_prefetch(&s[(base + (c+1)*16 + col)*4], 0, 1);

      // H1^T built directly in B-fragment layout, packed f16 fma + relu
      h2 x2 = h2{(_Float16)xs, (_Float16)xs};
      Frag16 Bh0, Bh1;
#pragma unroll
      for (int i = 0; i < 8; ++i) {
        h2 a0 = x2 * w1p[i]     + b1p[i];
        h2 a1 = x2 * w1p[8 + i] + b1p[8 + i];
        Bh0.p[i] = __builtin_elementwise_max(a0, hzero);
        Bh1.p[i] = __builtin_elementwise_max(a1, hzero);
      }

      // layer 2 transposed: 8x v_wmma_f32_16x16x32_f16, b2 folded into C init
      v8f acc[4];
#pragma unroll
      for (int mt = 0; mt < 4; ++mt) {
        v8f c0;
#pragma unroll
        for (int r = 0; r < 8; ++r) c0[r] = b2v[mt][r];
        c0 = __builtin_amdgcn_wmma_f32_16x16x32_f16(false, Af[0][mt].v, false, Bh0.v,
                                                    (short)0, c0, false, false);
        c0 = __builtin_amdgcn_wmma_f32_16x16x32_f16(false, Af[1][mt].v, false, Bh1.v,
                                                    (short)0, c0, false, false);
        acc[mt] = c0;
      }

      // layer 3: per-lane relu+dot over this lane's 32 channels, one xor-16 merge
      float t = 0.0f;
#pragma unroll
      for (int mt = 0; mt < 4; ++mt)
#pragma unroll
        for (int r = 0; r < 8; ++r)
          t = fmaf(relu32(acc[mt][r]), w3v[mt][r], t);

      t += __shfl_xor(t, 16, 32);

      float mval = 1.0f / (1.0f + __expf(-(t + b3v)));
      if (half == 0) mbuf[net][c*16 + col] = mval;   // sample index = col
    }

    __syncthreads();

    // cross-net min + 2-way softmax, one sample per thread
    int b = base + tid;
    if (b < B) {
      float m0 = mbuf[0][tid], m1 = mbuf[1][tid], m2 = mbuf[2][tid], m3 = mbuf[3][tid];
      float m4 = mbuf[4][tid], m5 = mbuf[5][tid], m6 = mbuf[6][tid], m7 = mbuf[7][tid];
      float s0 = fminf(fminf(m0, m1), fminf(m2, m3));
      float s1 = fminf(fminf(m4, m5), fminf(m6, m7));
      float z0 = 5.0f * s0, z1 = 5.0f * s1;
      float mx = fmaxf(z0, z1);
      float e0 = __expf(z0 - mx), e1 = __expf(z1 - mx);
      float inv = 1.0f / (e0 + e1);
      out[2*b]     = e0 * inv;
      out[2*b + 1] = e1 * inv;
    }

    __syncthreads();   // protect mbuf reuse on next tile
  }
}

extern "C" void kernel_launch(void* const* d_in, const int* in_sizes, int n_in,
                              void* d_out, int out_size, void* d_ws, size_t ws_size,
                              hipStream_t stream) {
  const float* s  = (const float*)d_in[0];
  const float* w1 = (const float*)d_in[1];
  const float* b1 = (const float*)d_in[2];
  const float* w2 = (const float*)d_in[3];
  const float* b2 = (const float*)d_in[4];
  const float* w3 = (const float*)d_in[5];
  const float* b3 = (const float*)d_in[6];
  float* out = (float*)d_out;

  int B = in_sizes[0] / 4;
  int tiles = (B + 255) / 256;
  int grid = tiles < 1024 ? tiles : 1024;   // grid-stride cap: ~8 waves/SIMD-class occupancy
  hipLaunchKernelGGL(mlp8_wmma_kernel, dim3(grid), dim3(256), 0, stream,
                     s, w1, b1, w2, b2, w3, b3, out, B);
}
